// LM_72069551227491
// MI455X (gfx1250) — compile-verified, matrix operand order checked by reading, not run on hardware
//
#include <hip/hip_runtime.h>
#include <math.h>

// ---------------------------------------------------------------------------
// Types for CDNA5 WMMA (wave32). bf16 stored as raw u16 everywhere; only the
// WMMA builtin sees the __bf16 vector type via a union bitcast.
// ---------------------------------------------------------------------------
typedef __attribute__((ext_vector_type(16))) __bf16 v16bf;
typedef __attribute__((ext_vector_type(8)))  float  v8f;
typedef __attribute__((ext_vector_type(4)))  unsigned int v4u;
typedef __attribute__((ext_vector_type(4)))  int v4i;

union Frag {
    v4u   q[2];   // two 16-byte chunks (8 bf16 each)
    v16bf v;      // fragment view for the builtin
};

__device__ __forceinline__ unsigned short f2bf(float f) {
    unsigned int u = __float_as_uint(f);
    unsigned int r = u + 0x7FFFu + ((u >> 16) & 1u);   // round-to-nearest-even
    return (unsigned short)(r >> 16);
}

// gfx1250 async copy global->LDS (ASYNCcnt-tracked). Guarded: if the builtin
// is not declared by this toolchain we fall back to direct global loads.
// Probe-confirmed prototype (from clang diagnostic):
//   void __builtin_amdgcn_global_load_async_to_lds_b128(
//       v4i addrspace(1)* gsrc, v4i addrspace(3)* ldst, imm int offset, imm int cpol)
#ifndef __has_builtin
#define __has_builtin(x) 0
#endif
#if __has_builtin(__builtin_amdgcn_global_load_async_to_lds_b128)
#define USE_ASYNC_LDS 1
typedef __attribute__((address_space(1))) v4i GV4;   // global 128-bit chunk
typedef __attribute__((address_space(3))) v4i LV4;   // LDS 128-bit chunk
#else
#define USE_ASYNC_LDS 0
#endif

// ---------------------------------------------------------------------------
// WMMA GEMM: C[M,N] = A[M,K] * B[N,K]^T  (+ bias[col]) (+ resid[row,col])
// A, B are bf16 (u16) row-major with leading dims lda/ldb (elements).
// Block: 256 threads = 8 waves, wave grid 4x2, each wave 32x32 (2x2 WMMA
// tiles) => block tile 128x64. M == gridDim.y*128, N == gridDim.x*64,
// K multiple of 32. Fragment layout per CDNA5 ISA 7.12.2:
//   A (16x32 bf16): lane l -> row M = l&15; K chunks at (l>>4)*8 and 16+(l>>4)*8
//   B (32x16 bf16): symmetric, lane l -> col N = l&15, same K pattern
//   C/D (16x16 f32): VGPR v -> M = v + 8*(l>>4), N = l&15
// B tile (64 rows x 32 K = 4KB) is staged in LDS via async copies, double
// buffered: the 4 waves in the M direction share it (4x VMEM traffic cut).
// A is read directly from global (only 2x redundancy, stays in WGP$/L2).
// ---------------------------------------------------------------------------
__global__ __launch_bounds__(256)
void wgemm_bf16(const unsigned short* __restrict__ A, int lda,
                const unsigned short* __restrict__ B, int ldb,
                float* __restrict__ C, int ldc, int K,
                const float* __restrict__ bias,
                const float* __restrict__ resid) {
    const int lane = threadIdx.x & 31;
    const int wave = threadIdx.x >> 5;
    const int wn = wave & 1;
    const int m0 = blockIdx.y * 128 + (wave >> 1) * 32;
    const int n0 = blockIdx.x * 64  + wn * 32;
    const int r  = lane & 15;
    const int hh = lane >> 4;

    const v8f vzero = {0.f,0.f,0.f,0.f,0.f,0.f,0.f,0.f};
    v8f acc00 = vzero, acc01 = vzero, acc10 = vzero, acc11 = vzero;

    const unsigned short* pa0 = A + (size_t)(m0 + r) * lda + hh * 8;
    const unsigned short* pa1 = pa0 + (size_t)16 * lda;

#if USE_ASYNC_LDS
    // ---- async-LDS staged B operand, double buffered ----
    __shared__ __align__(16) unsigned short Bs[2][64][32];
    const int crow = threadIdx.x >> 2;          // 0..63 : B row within tile
    const int cchk = (threadIdx.x & 3) * 8;     // 0,8,16,24 : K-element offset
    const unsigned short* gB = B + (size_t)(blockIdx.x * 64 + crow) * ldb + cchk;

    // preload buffer 0 (k = 0)
    __builtin_amdgcn_global_load_async_to_lds_b128(
        (GV4*)(size_t)gB, (LV4*)&Bs[0][crow][cchk], 0, 0);

    const int ns = K >> 5;
    for (int s = 0; s < ns; ++s) {
        const int k = s << 5;
        const int cur = s & 1;
        asm volatile("s_wait_asynccnt 0x0" ::: "memory"); // my async fills landed
        __syncthreads();                                  // everyone's landed; prev reads done
        if (s + 1 < ns) {
            __builtin_amdgcn_global_load_async_to_lds_b128(
                (GV4*)(size_t)(gB + k + 32), (LV4*)&Bs[cur ^ 1][crow][cchk], 0, 0);
        }
        Frag a0, a1, b0, b1;
        a0.q[0] = *(const v4u*)(pa0 + k);  a0.q[1] = *(const v4u*)(pa0 + k + 16);
        a1.q[0] = *(const v4u*)(pa1 + k);  a1.q[1] = *(const v4u*)(pa1 + k + 16);
        __builtin_prefetch(pa0 + k + 64, 0, 3);
        const unsigned short* lb0 = &Bs[cur][wn * 32 + r][hh * 8];
        const unsigned short* lb1 = &Bs[cur][wn * 32 + 16 + r][hh * 8];
        b0.q[0] = *(const v4u*)lb0;  b0.q[1] = *(const v4u*)(lb0 + 16);
        b1.q[0] = *(const v4u*)lb1;  b1.q[1] = *(const v4u*)(lb1 + 16);
        acc00 = __builtin_amdgcn_wmma_f32_16x16x32_bf16(false, a0.v, false, b0.v, (short)0, acc00, false, false);
        acc01 = __builtin_amdgcn_wmma_f32_16x16x32_bf16(false, a0.v, false, b1.v, (short)0, acc01, false, false);
        acc10 = __builtin_amdgcn_wmma_f32_16x16x32_bf16(false, a1.v, false, b0.v, (short)0, acc10, false, false);
        acc11 = __builtin_amdgcn_wmma_f32_16x16x32_bf16(false, a1.v, false, b1.v, (short)0, acc11, false, false);
    }
#else
    // ---- fallback: direct global fragment loads (known-good path) ----
    const unsigned short* pb0 = B + (size_t)(n0 + r) * ldb + hh * 8;
    const unsigned short* pb1 = pb0 + (size_t)16 * ldb;
    for (int k = 0; k < K; k += 32) {
        Frag a0, a1, b0, b1;
        a0.q[0] = *(const v4u*)(pa0 + k);  a0.q[1] = *(const v4u*)(pa0 + k + 16);
        a1.q[0] = *(const v4u*)(pa1 + k);  a1.q[1] = *(const v4u*)(pa1 + k + 16);
        b0.q[0] = *(const v4u*)(pb0 + k);  b0.q[1] = *(const v4u*)(pb0 + k + 16);
        b1.q[0] = *(const v4u*)(pb1 + k);  b1.q[1] = *(const v4u*)(pb1 + k + 16);
        __builtin_prefetch(pa0 + k + 64, 0, 3);
        __builtin_prefetch(pb0 + k + 64, 0, 3);
        acc00 = __builtin_amdgcn_wmma_f32_16x16x32_bf16(false, a0.v, false, b0.v, (short)0, acc00, false, false);
        acc01 = __builtin_amdgcn_wmma_f32_16x16x32_bf16(false, a0.v, false, b1.v, (short)0, acc01, false, false);
        acc10 = __builtin_amdgcn_wmma_f32_16x16x32_bf16(false, a1.v, false, b0.v, (short)0, acc10, false, false);
        acc11 = __builtin_amdgcn_wmma_f32_16x16x32_bf16(false, a1.v, false, b1.v, (short)0, acc11, false, false);
    }
#endif

    auto store_tile = [&](v8f acc, int mi, int nj) {
        const int col   = n0 + nj * 16 + r;
        const int rbase = m0 + mi * 16 + hh * 8;
        const float badd = bias ? bias[col] : 0.0f;
        #pragma unroll
        for (int v = 0; v < 8; ++v) {
            const size_t off = (size_t)(rbase + v) * ldc + col;
            float val = acc[v] + badd;
            if (resid) val += resid[off];
            C[off] = val;
        }
    };
    store_tile(acc00, 0, 0);
    store_tile(acc01, 0, 1);
    store_tile(acc10, 1, 0);
    store_tile(acc11, 1, 1);
}

// ---------------------------------------------------------------------------
// Embedding gather: x[t,d] = wte[idx[t], d]   (fp32)
// ---------------------------------------------------------------------------
__global__ void embed_k(const int* __restrict__ idx, const float* __restrict__ wte,
                        float* __restrict__ x, int n) {
    int i = blockIdx.x * blockDim.x + threadIdx.x;
    if (i < n) {
        int t = i / 768;
        int d = i - t * 768;
        x[i] = wte[(size_t)idx[t] * 768 + d];
    }
}

// fp32 -> bf16 elementwise
__global__ void cvt_bf16_k(const float* __restrict__ src, unsigned short* __restrict__ dst, int n) {
    int i = blockIdx.x * blockDim.x + threadIdx.x;
    if (i < n) dst[i] = f2bf(src[i]);
}

// exact GELU (erf) then bf16
__global__ void gelu_bf16_k(const float* __restrict__ src, unsigned short* __restrict__ dst, int n) {
    int i = blockIdx.x * blockDim.x + threadIdx.x;
    if (i < n) {
        float x = src[i];
        dst[i] = f2bf(0.5f * x * (1.0f + erff(x * 0.70710678118654752f)));
    }
}

// ---------------------------------------------------------------------------
// LayerNorm over D=768, output bf16. One block (256 threads) per row.
// ---------------------------------------------------------------------------
__global__ __launch_bounds__(256)
void ln_bf16_k(const float* __restrict__ X, const float* __restrict__ W,
               unsigned short* __restrict__ O) {
    __shared__ float red[256];
    const int t = blockIdx.x, tid = threadIdx.x;
    const float* row = X + (size_t)t * 768;
    float s = 0.f;
    for (int d = tid; d < 768; d += 256) s += row[d];
    red[tid] = s; __syncthreads();
    for (int o = 128; o > 0; o >>= 1) { if (tid < o) red[tid] += red[tid + o]; __syncthreads(); }
    const float mu = red[0] * (1.0f / 768.0f); __syncthreads();
    float v = 0.f;
    for (int d = tid; d < 768; d += 256) { float z = row[d] - mu; v += z * z; }
    red[tid] = v; __syncthreads();
    for (int o = 128; o > 0; o >>= 1) { if (tid < o) red[tid] += red[tid + o]; __syncthreads(); }
    const float rs = rsqrtf(red[0] * (1.0f / 768.0f) + 1e-5f);
    for (int d = tid; d < 768; d += 256)
        O[(size_t)t * 768 + d] = f2bf((row[d] - mu) * rs * W[d]);
}

// ---------------------------------------------------------------------------
// RoPE + head split. One block per token t, 384 threads = (head 0..11, i 0..31).
// Reads qkv fp32 [T,2304]; writes qb,kb bf16 [NH][T][64] and vT bf16 [NH][64][T].
// ---------------------------------------------------------------------------
__global__ __launch_bounds__(384)
void rope_split_k(const float* __restrict__ qkv,
                  unsigned short* __restrict__ qb,
                  unsigned short* __restrict__ kb,
                  unsigned short* __restrict__ vt) {
    const int t = blockIdx.x;
    const int head = threadIdx.x >> 5;   // 0..11
    const int i    = threadIdx.x & 31;   // 0..31
    const float inv = powf(10000.0f, -(float)(2 * i) / 64.0f);
    const float ang = (float)t * inv;
    const float c = cosf(ang), s = sinf(ang);
    const float* base = qkv + (size_t)t * 2304;

    const size_t qo = ((size_t)head * 2048 + t) * 64;
    float q1 = base[head * 64 + i];
    float q2 = base[head * 64 + 32 + i];
    qb[qo + i]      = f2bf(q1 * c - q2 * s);
    qb[qo + 32 + i] = f2bf(q1 * s + q2 * c);

    float k1 = base[768 + head * 64 + i];
    float k2 = base[768 + head * 64 + 32 + i];
    kb[qo + i]      = f2bf(k1 * c - k2 * s);
    kb[qo + 32 + i] = f2bf(k1 * s + k2 * c);

    float v1 = base[1536 + head * 64 + i];
    float v2 = base[1536 + head * 64 + 32 + i];
    vt[((size_t)head * 64 + i) * 2048 + t]      = f2bf(v1);
    vt[((size_t)head * 64 + i + 32) * 2048 + t] = f2bf(v2);
}

// ---------------------------------------------------------------------------
// Causal softmax row kernel: P[t,j] = softmax(S[t,:t+1]*0.125) (bf16), 0 past t.
// One block per row t.
// ---------------------------------------------------------------------------
__global__ __launch_bounds__(256)
void softmax_causal_k(const float* __restrict__ S, unsigned short* __restrict__ P) {
    __shared__ float red[256];
    const int t = blockIdx.x, tid = threadIdx.x;
    const float* row = S + (size_t)t * 2048;
    unsigned short* prow = P + (size_t)t * 2048;
    const float scale = 0.125f;   // 1/sqrt(64)
    float mx = -3.0e38f;
    for (int j = tid; j <= t; j += 256) mx = fmaxf(mx, row[j] * scale);
    red[tid] = mx; __syncthreads();
    for (int o = 128; o > 0; o >>= 1) { if (tid < o) red[tid] = fmaxf(red[tid], red[tid + o]); __syncthreads(); }
    mx = red[0]; __syncthreads();
    float sum = 0.f;
    for (int j = tid; j <= t; j += 256) sum += __expf(row[j] * scale - mx);
    red[tid] = sum; __syncthreads();
    for (int o = 128; o > 0; o >>= 1) { if (tid < o) red[tid] += red[tid + o]; __syncthreads(); }
    const float inv = 1.0f / red[0];
    for (int j = tid; j < 2048; j += 256)
        prow[j] = (j <= t) ? f2bf(__expf(row[j] * scale - mx) * inv) : (unsigned short)0;
}

// ---------------------------------------------------------------------------
// Host-side orchestration
// ---------------------------------------------------------------------------
extern "C" void kernel_launch(void* const* d_in, const int* in_sizes, int n_in,
                              void* d_out, int out_size, void* d_ws, size_t ws_size,
                              hipStream_t stream) {
    (void)in_sizes; (void)n_in; (void)out_size; (void)ws_size;

    const int*   idx        = (const int*)  d_in[0];
    const float* wte        = (const float*)d_in[1];
    const float* ln1_w      = (const float*)d_in[2];
    const float* attn_w     = (const float*)d_in[3];   // [L][2304][768]
    const float* attnproj_w = (const float*)d_in[4];   // [L][768][768]
    const float* ln2_w      = (const float*)d_in[5];
    const float* fc_w       = (const float*)d_in[6];   // [L][3072][768]
    const float* fc_b       = (const float*)d_in[7];   // [L][3072]
    const float* proj_w     = (const float*)d_in[8];   // [L][768][3072]
    const float* proj_b     = (const float*)d_in[9];   // [L][768]
    const float* lnf_w      = (const float*)d_in[10];
    const float* unemb_b    = (const float*)d_in[11];  // [V]
    float* out = (float*)d_out;

    constexpr int T = 2048, D = 768, NH = 12, HD = 64, L = 4, V = 32000;

    // ---- workspace partition (256B aligned chunks) ----
    char* ws = (char*)d_ws;
    size_t off = 0;
    auto take = [&](size_t bytes) -> char* {
        char* p = ws + off;
        off += (bytes + 255) & ~(size_t)255;
        return p;
    };
    float*          x    = (float*)         take((size_t)T * D * 4);       // residual stream
    unsigned short* hbf  = (unsigned short*)take((size_t)T * 4 * D * 2);   // bf16 activation (LN out / y / m)
    float*          qkv  = (float*)         take((size_t)T * 3 * D * 4);
    unsigned short* qb   = (unsigned short*)take((size_t)NH * T * HD * 2);
    unsigned short* kb   = (unsigned short*)take((size_t)NH * T * HD * 2);
    unsigned short* vt   = (unsigned short*)take((size_t)NH * HD * T * 2); // V transposed [NH][HD][T]
    float*          y    = (float*)         take((size_t)T * D * 4);       // attention output
    float*          sbuf = (float*)         take((size_t)T * T * 4);       // scores (per head)
    unsigned short* pbuf = (unsigned short*)take((size_t)T * T * 2);       // probs bf16 (per head)
    unsigned short* wb   = (unsigned short*)take((size_t)4 * D * D * 2);   // bf16 weight scratch
    unsigned short* wteb = (unsigned short*)take((size_t)V * D * 2);       // bf16 wte for unembedding
    float* fco = sbuf;  // fc activations [T,3072] fp32 overlay (sbuf+pbuf == exactly T*4D*4 bytes)

    auto cvt = [&](const float* s, unsigned short* d, int n) {
        cvt_bf16_k<<<(n + 255) / 256, 256, 0, stream>>>(s, d, n);
    };
    auto gemm = [&](const unsigned short* A, int lda, const unsigned short* B, int ldb,
                    float* C, int ldc, int N, int K, const float* bias, const float* resid) {
        dim3 grid(N / 64, T / 128);
        wgemm_bf16<<<grid, 256, 0, stream>>>(A, lda, B, ldb, C, ldc, K, bias, resid);
    };

    // ---- embedding + one-time wte bf16 conversion ----
    embed_k<<<(T * D + 255) / 256, 256, 0, stream>>>(idx, wte, x, T * D);
    cvt(wte, wteb, V * D);

    for (int l = 0; l < L; ++l) {
        // LN1 -> bf16
        ln_bf16_k<<<T, 256, 0, stream>>>(x, ln1_w + (size_t)l * D, hbf);
        // qkv = h @ attn_w^T    (M=2048, N=2304, K=768)
        cvt(attn_w + (size_t)l * 3 * D * D, wb, 3 * D * D);
        gemm(hbf, D, wb, D, qkv, 3 * D, 3 * D, D, nullptr, nullptr);
        // RoPE + split into heads (bf16)
        rope_split_k<<<T, 384, 0, stream>>>(qkv, qb, kb, vt);
        // attention per head: S = Q K^T ; P = causal softmax ; Y_h = P V
        for (int h = 0; h < NH; ++h) {
            const unsigned short* qh = qb + (size_t)h * T * HD;
            const unsigned short* kh = kb + (size_t)h * T * HD;
            const unsigned short* vh = vt + (size_t)h * HD * T;
            gemm(qh, HD, kh, HD, sbuf, T, T, HD, nullptr, nullptr);          // M=T,N=T,K=64
            softmax_causal_k<<<T, 256, 0, stream>>>(sbuf, pbuf);
            gemm(pbuf, T, vh, T, y + h * HD, D, HD, T, nullptr, nullptr);    // M=T,N=64,K=T
        }
        // x = x + y @ attnproj_w^T
        cvt(y, hbf, T * D);
        cvt(attnproj_w + (size_t)l * D * D, wb, D * D);
        gemm(hbf, D, wb, D, x, D, D, D, nullptr, x);
        // LN2 -> bf16
        ln_bf16_k<<<T, 256, 0, stream>>>(x, ln2_w + (size_t)l * D, hbf);
        // fc: fco = h @ fc_w^T + fc_b   (N=3072, K=768)
        cvt(fc_w + (size_t)l * 4 * D * D, wb, 4 * D * D);
        gemm(hbf, D, wb, D, fco, 4 * D, 4 * D, D, fc_b + (size_t)l * 4 * D, nullptr);
        // GELU -> bf16
        gelu_bf16_k<<<(T * 4 * D + 255) / 256, 256, 0, stream>>>(fco, hbf, T * 4 * D);
        // x = x + m @ proj_w^T + proj_b  (N=768, K=3072)
        cvt(proj_w + (size_t)l * 4 * D * D, wb, 4 * D * D);
        gemm(hbf, 4 * D, wb, 4 * D, x, D, D, 4 * D, proj_b + (size_t)l * D, x);
    }

    // final LN + unembedding: out = lnf(x) @ wte^T + unemb_b  (N=32000, K=768)
    ln_bf16_k<<<T, 256, 0, stream>>>(x, lnf_w, hbf);
    gemm(hbf, D, wteb, D, out, V, V, D, unemb_b, nullptr);
}
